// CNN_window_self_attention_15126874817245
// MI455X (gfx1250) — matrix operand order repaired and checked by reading, not compiled, not run on hardware
//
#include <hip/hip_runtime.h>
#include <hip/hip_bf16.h>
#include <math.h>

// ---------------------------------------------------------------------------
// Swin window attention for MI455X (gfx1250, wave32, WMMA f32_16x16x32_f16)
//   B_=2048 windows, N=49 tokens (padded to 64), D=384, H=12, hd=32, NW=64
// Pipeline: cvt weights -> qkv GEMM -> per-(b,h) attention (TDM-fed) -> proj
// ---------------------------------------------------------------------------

typedef __attribute__((ext_vector_type(16))) _Float16 v16h;
typedef __attribute__((ext_vector_type(8)))  _Float16 v8h;
typedef __attribute__((ext_vector_type(8)))  float    v8f;
typedef __attribute__((ext_vector_type(4)))  unsigned v4u;
typedef __attribute__((ext_vector_type(4)))  int      v4i;
typedef __attribute__((ext_vector_type(8)))  int      v8i;

#define B_WIN   2048
#define NTOK    49
#define NPAD    64
#define DIM     384
#define NH      12
#define HD      32
#define QKVDIM  1152

// padded LDS strides (halfs) chosen for conflict-free b128 fragment reads
#define QK_STRIDE 40   // 32 data + 4-DWORD TDM pad : 80B = 20 banks
#define VT_STRIDE 72   // 64 data + 4-DWORD TDM pad : 144B = 36 banks

#if __has_builtin(__builtin_amdgcn_tensor_load_to_lds)
#define HAS_TDM 1
#else
#define HAS_TDM 0
#endif

// -------------------- WMMA fragment loaders (ISA 7.12.2 layouts) -----------

// A 16x32 f16: lanes 0-15 row r: elems0-7 = K k0..k0+7, elems8-15 = K k0+16..23
//              lanes 16-31 row r: elems0-7 = K k0+8..15, elems8-15 = K k0+24..31
static __device__ inline v16h frag_a(const _Float16* base, int stride,
                                     int lane, int row0, int k0) {
  int r  = lane & 15;
  int kb = k0 + ((lane >> 4) << 3);
  const _Float16* p = base + (row0 + r) * stride + kb;
  v8h lo = *(const v8h*)(p);
  v8h hi = *(const v8h*)(p + 16);
  return __builtin_shufflevector(lo, hi, 0,1,2,3,4,5,6,7,8,9,10,11,12,13,14,15);
}

// B 32x16 f16 from transposed source BT[N][K]: lane holds col n, K in same order as frag_a
static __device__ inline v16h frag_b(const _Float16* bt, int stride,
                                     int lane, int n0, int k0) {
  int c  = n0 + (lane & 15);
  int kb = k0 + ((lane >> 4) << 3);
  const _Float16* p = bt + c * stride + kb;
  v8h lo = *(const v8h*)(p);
  v8h hi = *(const v8h*)(p + 16);
  return __builtin_shufflevector(lo, hi, 0,1,2,3,4,5,6,7,8,9,10,11,12,13,14,15);
}

static __device__ inline v8f wmma_f16(v16h a, v16h b, v8f c) {
  return __builtin_amdgcn_wmma_f32_16x16x32_f16(false, a, false, b,
                                                (short)0, c, false, false);
}

// 16-lane XOR-tree reductions via ds_swizzle (and=0x1F, xor stays in 16-group)
static __device__ inline float red_max16(float v) {
  float t;
  t = __int_as_float(__builtin_amdgcn_ds_swizzle(__float_as_int(v), 0x1F | (1 << 10))); v = fmaxf(v, t);
  t = __int_as_float(__builtin_amdgcn_ds_swizzle(__float_as_int(v), 0x1F | (2 << 10))); v = fmaxf(v, t);
  t = __int_as_float(__builtin_amdgcn_ds_swizzle(__float_as_int(v), 0x1F | (4 << 10))); v = fmaxf(v, t);
  t = __int_as_float(__builtin_amdgcn_ds_swizzle(__float_as_int(v), 0x1F | (8 << 10))); v = fmaxf(v, t);
  return v;
}
static __device__ inline float red_sum16(float v) {
  v += __int_as_float(__builtin_amdgcn_ds_swizzle(__float_as_int(v), 0x1F | (1 << 10)));
  v += __int_as_float(__builtin_amdgcn_ds_swizzle(__float_as_int(v), 0x1F | (2 << 10)));
  v += __int_as_float(__builtin_amdgcn_ds_swizzle(__float_as_int(v), 0x1F | (4 << 10)));
  v += __int_as_float(__builtin_amdgcn_ds_swizzle(__float_as_int(v), 0x1F | (8 << 10)));
  return v;
}

// -------------------- TDM: 1-D f16 tile, global -> LDS, with row padding ---
// D# per CDNA5 ISA: group0 {count=1, lds_addr, global_addr,
// type=2}; group1 {data_size=2B, pad_enable, pad_interval, pad_amount,
// tensor_dim0=tile_dim0=nhalfs (1 row)}. Pad inserts (pamt+1) DWORDs every
// 2^(pint+1) DWORDs -> builds the padded LDS strides above during the DMA.
#if HAS_TDM
static __device__ inline void tdm_load_f16(void* lds_ptr, const _Float16* gptr,
                                           unsigned nhalfs, unsigned pint,
                                           unsigned pamt) {
  unsigned long long ga = (unsigned long long)(const void*)gptr;
  unsigned lds_off = (unsigned)(unsigned long long)lds_ptr;   // LDS aperture offset
  v4u g0;
  g0[0] = 1u;                                                 // count=1 valid D#
  g0[1] = lds_off;                                            // lds_addr
  g0[2] = (unsigned)(ga & 0xFFFFFFFFu);                       // global_addr lo
  g0[3] = (unsigned)((ga >> 32) & 0x01FFFFFFu) | (2u << 30);  // global hi | type=2
  v8i g1;
  g1[0] = (int)((1u << 16) | (1u << 20) | (pint << 22) | (pamt << 25));
  g1[1] = (int)((nhalfs & 0xFFFFu) << 16);                    // tensor_dim0 lo16
  g1[2] = (int)(((nhalfs >> 16) & 0xFFFFu)) | (1 << 16);      // dim0 hi | tensor_dim1=1
  g1[3] = (int)((nhalfs & 0xFFFFu) << 16);                    // tile_dim0
  g1[4] = 0;                                                  // tile_dim1/2 unused
  g1[5] = (int)nhalfs;                                        // dim0_stride (unused, 1 row)
  g1[6] = 0;
  g1[7] = 0;
  v4i z4 = (v4i){0, 0, 0, 0};
#if __clang_major__ >= 23
  v8i z8 = (v8i){0, 0, 0, 0, 0, 0, 0, 0};
  __builtin_amdgcn_tensor_load_to_lds(g0, g1, z4, z4, z8, 0);
#else
  __builtin_amdgcn_tensor_load_to_lds(g0, g1, z4, z4, 0);
#endif
}
#endif

// -------------------- kernel 0: fp32 -> f16 weight conversion --------------
__global__ __launch_bounds__(256) void wsa_cvt_weights(
    const float* __restrict__ qkvw, const float* __restrict__ projw,
    _Float16* __restrict__ w16q, _Float16* __restrict__ w16p) {
  int i = blockIdx.x * 256 + threadIdx.x;
  if (i < QKVDIM * DIM) w16q[i] = (_Float16)qkvw[i];
  if (i < DIM * DIM)    w16p[i] = (_Float16)projw[i];
}

// -------------------- kernel 1: qkv = x @ qkv_w.T + b ----------------------
// grid (2048, 18) : (window, 64-out-channel tile). 4 waves, wave w = M-tile w.
__global__ __launch_bounds__(128) void wsa_qkv_gemm(
    const float* __restrict__ x, const float* __restrict__ qkvb,
    const _Float16* __restrict__ w16,
    _Float16* __restrict__ qg, _Float16* __restrict__ kg,
    _Float16* __restrict__ vtg) {
  __shared__ _Float16 xs[NPAD * 392];           // x window as f16, zero-padded rows
  const int b = blockIdx.x, ct = blockIdx.y;
  const int tid = threadIdx.x, lane = tid & 31, wv = tid >> 5;

  const float* xb = x + (size_t)b * (NTOK * DIM);
  for (int i = tid; i < NPAD * DIM; i += 128) {
    int r = i / DIM, c = i - r * DIM;
    float v = (r < NTOK) ? xb[r * DIM + c] : 0.f;
    xs[r * 392 + c] = (_Float16)v;
  }
  __syncthreads();

  const int c0 = ct * 64;
  v8f acc[4];
#pragma unroll
  for (int nt = 0; nt < 4; ++nt) acc[nt] = (v8f){0,0,0,0,0,0,0,0};

  for (int ks = 0; ks < DIM / 32; ++ks) {
    int k0 = ks * 32;
    v16h a = frag_a(xs, 392, lane, wv * 16, k0);
    if (ks + 2 < DIM / 32)                       // stream next-next weight slab
      __builtin_prefetch(w16 + (c0 + (lane & 15)) * DIM + k0 + 64, 0, 0);
#pragma unroll
    for (int nt = 0; nt < 4; ++nt) {
      v16h bf = frag_b(w16, DIM, lane, c0 + nt * 16, k0);
      acc[nt] = wmma_f16(a, bf, acc[nt]);
    }
  }

  const int mbase = wv * 16 + ((lane >> 4) << 3);
  const float qscale = 0.17677669529663689f;    // hd^-0.5
#pragma unroll
  for (int nt = 0; nt < 4; ++nt) {
    int c = c0 + nt * 16 + (lane & 15);
    float bias = qkvb[c];
#pragma unroll
    for (int vr = 0; vr < 8; ++vr) {
      int row = mbase + vr;
      float val = (row < NTOK) ? (acc[nt][vr] + bias) : 0.f;  // zero the pad rows
      if (c < DIM) {
        int h = c >> 5, d = c & 31;
        qg[(((size_t)b * NH + h) * NPAD + row) * HD + d] = (_Float16)(val * qscale);
      } else if (c < 2 * DIM) {
        int cc = c - DIM; int h = cc >> 5, d = cc & 31;
        kg[(((size_t)b * NH + h) * NPAD + row) * HD + d] = (_Float16)val;
      } else {
        int cc = c - 2 * DIM; int h = cc >> 5, d = cc & 31;
        vtg[(((size_t)b * NH + h) * HD + d) * NPAD + row] = (_Float16)val;  // v^T
      }
    }
  }
}

// -------------------- kernel 2: attention per (b, h) -----------------------
// grid (2048, 12), 4 waves. TDM DMAs q/k/v^T into padded LDS while the waves
// stage bias+mask; S = q k^T (one K=32 WMMA/tile), register softmax via
// ds_swizzle, O = P V (K=64, 2 WMMA steps).
__global__ __launch_bounds__(128) void wsa_attn(
    const _Float16* __restrict__ qg, const _Float16* __restrict__ kg,
    const _Float16* __restrict__ vtg, const float* __restrict__ rpb,
    const float* __restrict__ mask, _Float16* __restrict__ ao) {
  __shared__ float    bl[NPAD * NPAD];          // bias + mask (f32)
  __shared__ _Float16 pl[NPAD * 72];            // softmax probs (f16)
  __shared__ _Float16 qs[NPAD * QK_STRIDE];     // q tile  (TDM, row-padded)
  __shared__ _Float16 ks_[NPAD * QK_STRIDE];    // k tile
  __shared__ _Float16 vs[HD * VT_STRIDE];       // v^T tile
  const int b = blockIdx.x, h = blockIdx.y;
  const int tid = threadIdx.x, lane = tid & 31, wv = tid >> 5;

  const _Float16* qb = qg  + ((size_t)b * NH + h) * (NPAD * HD);
  const _Float16* kb = kg  + ((size_t)b * NH + h) * (NPAD * HD);
  const _Float16* vb = vtg + ((size_t)b * NH + h) * (HD * NPAD);

#if HAS_TDM
  if (wv == 0) {      // one wave issues the 3 DMAs (4KB each, contiguous src)
    tdm_load_f16(qs,  qb, NPAD * HD, 3, 3);   // pad 4 DW every 16 DW row
    tdm_load_f16(ks_, kb, NPAD * HD, 3, 3);
    tdm_load_f16(vs,  vb, HD * NPAD, 4, 3);   // pad 4 DW every 32 DW row
  }
#else
  for (int i = tid; i < NPAD * HD; i += 128) {
    qs [(i >> 5) * QK_STRIDE + (i & 31)] = qb[i];
    ks_[(i >> 5) * QK_STRIDE + (i & 31)] = kb[i];
    vs [(i >> 6) * VT_STRIDE + (i & 63)] = vb[i];
  }
#endif

  const float* mw = mask + (size_t)(b & 63) * (NTOK * NTOK);
  for (int i = tid; i < NPAD * NPAD; i += 128) {
    int r = i >> 6, c = i & 63;
    float v = -1e30f;                            // masks pad cols j>=49
    if (r < NTOK && c < NTOK) {
      int yi = r / 7, xi = r % 7, yj = c / 7, xj = c % 7;
      int idx = (yi - yj + 6) * 13 + (xi - xj + 6);   // relative position index
      v = rpb[idx * NH + h] + mw[r * NTOK + c];
    }
    bl[i] = v;
  }
#if HAS_TDM
  if (wv == 0) __builtin_amdgcn_s_wait_tensorcnt(0);
#endif
  __syncthreads();

  v8f st[4];
  {
    v16h a = frag_a(qs, QK_STRIDE, lane, wv * 16, 0);
#pragma unroll
    for (int nt = 0; nt < 4; ++nt) {
      v16h bf = frag_b(ks_, QK_STRIDE, lane, nt * 16, 0);
      st[nt] = wmma_f16(a, bf, (v8f){0,0,0,0,0,0,0,0});
    }
  }

  const int mbase = wv * 16 + ((lane >> 4) << 3);
  const int ncol  = lane & 15;
#pragma unroll
  for (int nt = 0; nt < 4; ++nt)
#pragma unroll
    for (int vr = 0; vr < 8; ++vr)
      st[nt][vr] += bl[(mbase + vr) * NPAD + nt * 16 + ncol];

  // softmax: each row's 64 entries = one VGPR slot across 4 tiles x 16 lanes
#pragma unroll
  for (int vr = 0; vr < 8; ++vr) {
    float mx = st[0][vr];
#pragma unroll
    for (int nt = 1; nt < 4; ++nt) mx = fmaxf(mx, st[nt][vr]);
    mx = red_max16(mx);
    float s = 0.f;
#pragma unroll
    for (int nt = 0; nt < 4; ++nt) {
      float e = __expf(st[nt][vr] - mx);
      st[nt][vr] = e; s += e;
    }
    s = red_sum16(s);
    float rs = __frcp_rn(s);
#pragma unroll
    for (int nt = 0; nt < 4; ++nt) st[nt][vr] *= rs;
  }

  // P to LDS f16 (each wave touches only its own 16 rows -> no barrier)
#pragma unroll
  for (int nt = 0; nt < 4; ++nt)
#pragma unroll
    for (int vr = 0; vr < 8; ++vr)
      pl[(mbase + vr) * 72 + nt * 16 + ncol] = (_Float16)st[nt][vr];

  v8f oa[2];
  oa[0] = (v8f){0,0,0,0,0,0,0,0};
  oa[1] = (v8f){0,0,0,0,0,0,0,0};
#pragma unroll
  for (int ks2 = 0; ks2 < 2; ++ks2) {
    v16h ap = frag_a(pl, 72, lane, wv * 16, ks2 * 32);
#pragma unroll
    for (int nt = 0; nt < 2; ++nt) {
      v16h bv = frag_b(vs, VT_STRIDE, lane, nt * 16, ks2 * 32);
      oa[nt] = wmma_f16(ap, bv, oa[nt]);
    }
  }

#pragma unroll
  for (int nt = 0; nt < 2; ++nt)
#pragma unroll
    for (int vr = 0; vr < 8; ++vr) {
      int row = mbase + vr;
      if (row < NTOK)
        ao[((size_t)b * NTOK + row) * DIM + h * HD + nt * 16 + ncol] =
            (_Float16)oa[nt][vr];
    }
}

// -------------------- kernel 3: out = ao @ proj_w.T (fp32 out) -------------
// grid (2048, 6) : (window, 64-channel tile)
__global__ __launch_bounds__(128) void wsa_proj_gemm(
    const _Float16* __restrict__ ao, const _Float16* __restrict__ w16,
    float* __restrict__ out) {
  const int b = blockIdx.x, ct = blockIdx.y;
  const int tid = threadIdx.x, lane = tid & 31, wv = tid >> 5;
  const int c0 = ct * 64;
  const _Float16* ab = ao + (size_t)b * NTOK * DIM;

  v8f acc[4];
#pragma unroll
  for (int nt = 0; nt < 4; ++nt) acc[nt] = (v8f){0,0,0,0,0,0,0,0};

  for (int ks = 0; ks < DIM / 32; ++ks) {
    int k0 = ks * 32;
    v16h a = {};
    {
      int r = lane & 15, row = wv * 16 + r;
      int kb = k0 + ((lane >> 4) << 3);
      if (row < NTOK) {
        const _Float16* p = ab + row * DIM + kb;
        v8h lo = *(const v8h*)(p);
        v8h hi = *(const v8h*)(p + 16);
        a = __builtin_shufflevector(lo, hi, 0,1,2,3,4,5,6,7,8,9,10,11,12,13,14,15);
      }
    }
    if (ks + 2 < DIM / 32)
      __builtin_prefetch(w16 + (c0 + (lane & 15)) * DIM + k0 + 64, 0, 0);
#pragma unroll
    for (int nt = 0; nt < 4; ++nt) {
      v16h bf = frag_b(w16, DIM, lane, c0 + nt * 16, k0);
      acc[nt] = wmma_f16(a, bf, acc[nt]);
    }
  }

  const int mbase = wv * 16 + ((lane >> 4) << 3);
#pragma unroll
  for (int nt = 0; nt < 4; ++nt)
#pragma unroll
    for (int vr = 0; vr < 8; ++vr) {
      int row = mbase + vr;
      if (row < NTOK)
        out[((size_t)b * NTOK + row) * DIM + c0 + nt * 16 + (lane & 15)] =
            acc[nt][vr];
    }
}

// -------------------- host launch ------------------------------------------
extern "C" void kernel_launch(void* const* d_in, const int* in_sizes, int n_in,
                              void* d_out, int out_size, void* d_ws, size_t ws_size,
                              hipStream_t stream) {
  const float* x     = (const float*)d_in[0];
  const float* mask  = (const float*)d_in[1];
  const float* qkvw  = (const float*)d_in[2];
  const float* qkvb  = (const float*)d_in[3];
  const float* projw = (const float*)d_in[4];
  const float* rpb   = (const float*)d_in[5];
  float* out = (float*)d_out;

  // workspace layout (bytes, all 256-aligned)
  char* ws = (char*)d_ws;
  const size_t SZ_W16Q = (size_t)QKVDIM * DIM * 2;            //    884,736
  const size_t SZ_W16P = (size_t)DIM * DIM * 2;               //    294,912
  const size_t SZ_QK   = (size_t)B_WIN * NH * NPAD * HD * 2;  // 100,663,296
  const size_t SZ_AO   = (size_t)B_WIN * NTOK * DIM * 2;      //  77,070,336
  _Float16* w16q = (_Float16*)(ws);
  _Float16* w16p = (_Float16*)(ws + SZ_W16Q);
  _Float16* qg   = (_Float16*)(ws + SZ_W16Q + SZ_W16P);
  _Float16* kg   = (_Float16*)((char*)qg + SZ_QK);
  _Float16* vtg  = (_Float16*)((char*)kg + SZ_QK);
  _Float16* ao   = (_Float16*)((char*)vtg + SZ_QK);
  (void)ao; (void)SZ_AO; (void)ws_size; (void)in_sizes; (void)n_in; (void)out_size;

  wsa_cvt_weights<<<(QKVDIM * DIM + 255) / 256, 256, 0, stream>>>(qkvw, projw, w16q, w16p);
  wsa_qkv_gemm<<<dim3(B_WIN, QKVDIM / 64), 128, 0, stream>>>(x, qkvb, w16q, qg, kg, vtg);
  wsa_attn<<<dim3(B_WIN, NH), 128, 0, stream>>>(qg, kg, vtg, rpb, mask, ao);
  wsa_proj_gemm<<<dim3(B_WIN, DIM / 64), 128, 0, stream>>>(ao, w16p, out);
}
